// MultiHeadAttention_30932354465901
// MI455X (gfx1250) — matrix operand order repaired and checked by reading, not compiled, not run on hardware
//
#include <hip/hip_runtime.h>
#include <hip/hip_bf16.h>

typedef _Float16 h16;
typedef __attribute__((ext_vector_type(16))) _Float16 v16h;
typedef __attribute__((ext_vector_type(8)))  _Float16 v8h;
typedef __attribute__((ext_vector_type(8)))  float    v8f;

#define DM   1024   // d_model
#define NH   16     // heads
#define DKH  64     // d_k per head
#define BB   2      // batch
#define SS   2048   // seq
#define MTOT (BB*SS)  // 4096 rows

// ---- WMMA fragment helpers (contiguous k order, not the ISA table order, wave32) ----

// A-matrix 16x32 f16: lane m = lane&15, half = lane>>4.
// element j=2v+w  ->  K = 2v + 16*half + w
__device__ __forceinline__ int a_koff(int v, int half) {
  return 2 * v + 16 * half;
}

__device__ __forceinline__ v16h load_a_f16(const h16* __restrict__ row, int k0, int half) {
  v16h a;
#pragma unroll
  for (int v = 0; v < 8; ++v) {
    int k = k0 + a_koff(v, half);
    a[2 * v]     = row[k];
    a[2 * v + 1] = row[k + 1];
  }
  return a;
}

// B-matrix 32x16 f16: lane n = lane&15, K = 16*(lane>>4) + j, j contiguous in memory.
__device__ __forceinline__ v16h load_b_f16(const h16* __restrict__ p) {
  return *reinterpret_cast<const v16h*>(p);
}

__device__ __forceinline__ v8f wmma16(v16h a, v16h b, v8f c) {
  return __builtin_amdgcn_wmma_f32_16x16x32_f16(false, a, false, b, (short)0, c, false, false);
}

// ====================================================================
// Kernel 0: one-shot fp32 -> f16 streaming conversion (8 elems/thread)
// ====================================================================
__global__ __launch_bounds__(256) void cvt_f32_f16_kernel(
    const float* __restrict__ in, h16* __restrict__ out, int n8)
{
  const int i = blockIdx.x * 256 + threadIdx.x;
  if (i >= n8) return;
  const float4* p = reinterpret_cast<const float4*>(in) + (size_t)i * 2;
  const float4 a = p[0];
  const float4 b = p[1];
  v8h o;
  o[0] = (_Float16)a.x; o[1] = (_Float16)a.y; o[2] = (_Float16)a.z; o[3] = (_Float16)a.w;
  o[4] = (_Float16)b.x; o[5] = (_Float16)b.y; o[6] = (_Float16)b.z; o[7] = (_Float16)b.w;
  *(reinterpret_cast<v8h*>(out) + i) = o;
}

// ====================================================================
// Kernel 1: projection  Y = X @ W^T + b  (f16 in, f16 out, f32 accum)
// vmode 0: out[b][h][s][d]   (Q, K head-major)
// vmode 1: out[b][h][d][s]   (V transposed for PV B-fragments)
// One wave -> 16(M) x 64(N) tile. 4096 waves total.
// ====================================================================
__global__ __launch_bounds__(256) void mha_proj_kernel(
    const h16* __restrict__ X, const h16* __restrict__ W,
    const float* __restrict__ bias, h16* __restrict__ Out, int vmode)
{
  const int wid  = threadIdx.x >> 5;
  const int lane = threadIdx.x & 31;
  const int gw   = blockIdx.x * 8 + wid;
  const int tileN = gw & 15;    // 16 col-tiles of 64
  const int tileM = gw >> 4;    // 256 row-tiles of 16
  const int half = lane >> 4;
  const int ln   = lane & 15;

  const h16* xrow = X + (size_t)(tileM * 16 + ln) * DM;

  v8f acc[4];
#pragma unroll
  for (int i = 0; i < 4; ++i)
#pragma unroll
    for (int r = 0; r < 8; ++r) acc[i][r] = 0.0f;

  for (int k0 = 0; k0 < DM; k0 += 32) {
    __builtin_prefetch(xrow + k0 + 64, 0, 0);
    v16h a = load_a_f16(xrow, k0, half);
#pragma unroll
    for (int nb = 0; nb < 4; ++nb) {
      const int wrow = tileN * 64 + nb * 16 + ln;
      v16h b = load_b_f16(W + (size_t)wrow * DM + k0 + half * 16);
      acc[nb] = wmma16(a, b, acc[nb]);
    }
  }

#pragma unroll
  for (int nb = 0; nb < 4; ++nb) {
    const int ncol = tileN * 64 + nb * 16 + ln;
    const float bv = bias[ncol];
    const int hh = ncol >> 6;     // head
    const int d  = ncol & 63;     // dim within head
#pragma unroll
    for (int r = 0; r < 8; ++r) {
      const int gm = tileM * 16 + r + 8 * half;   // global row
      const int bi = gm >> 11;                     // batch (gm / 2048)
      const int s  = gm & (SS - 1);
      const float val = acc[nb][r] + bv;
      size_t off;
      if (vmode) off = (((size_t)(bi * NH + hh)) * DKH + d) * SS + s;
      else       off = (((size_t)(bi * NH + hh)) * SS + s) * DKH + d;
      Out[off] = (h16)val;
    }
  }
}

// ====================================================================
// Kernel 2: causal flash attention, one wave per (b, h, 16-query tile)
// Q,K: [B,H,S,64] f16   V: [B,H,64,S] f16   O: [B,S,D] f16
// ====================================================================
__global__ __launch_bounds__(256) void mha_attn_kernel(
    const h16* __restrict__ Qh, const h16* __restrict__ Kh,
    const h16* __restrict__ Vt, h16* __restrict__ Oflat)
{
  __shared__ _Float16 lds_p[8][16 * 32];   // P tile per wave (C->A transpose)

  const int wid  = threadIdx.x >> 5;
  const int lane = threadIdx.x & 31;
  const int gw   = blockIdx.x * 8 + wid;
  const int qt = gw & 127;                // S/16 = 128 query tiles
  const int hh = (gw >> 7) & 15;
  const int bi = gw >> 11;
  const int q0 = qt * 16;
  const int half = lane >> 4;
  const int ln   = lane & 15;

  const h16* qbase = Qh + ((size_t)(bi * NH + hh)) * SS * DKH;
  const h16* kbase = Kh + ((size_t)(bi * NH + hh)) * SS * DKH;
  const h16* vbase = Vt + ((size_t)(bi * NH + hh)) * DKH * SS;

  // Q A-fragments (dk = 64 -> two K=32 chunks), loaded once
  const h16* qrow = qbase + (size_t)(q0 + ln) * DKH;
  v16h aq0 = load_a_f16(qrow, 0,  half);
  v16h aq1 = load_a_f16(qrow, 32, half);

  v8f o[4];
  float mi[8], li[8];
#pragma unroll
  for (int nb = 0; nb < 4; ++nb)
#pragma unroll
    for (int r = 0; r < 8; ++r) o[nb][r] = 0.0f;
#pragma unroll
  for (int r = 0; r < 8; ++r) { mi[r] = -3.0e38f; li[r] = 0.0f; }

  const float scale = 0.125f;   // 1/sqrt(64)

  for (int j0 = 0; j0 < q0 + 16; j0 += 32) {
    // ---- scores S = Q K^T for keys [j0, j0+32) : two 16x16 f32 blocks ----
    v8f sc[2];
#pragma unroll
    for (int nb = 0; nb < 2; ++nb) {
      const h16* krow = kbase + (size_t)(j0 + nb * 16 + ln) * DKH;
      v16h bk0 = load_b_f16(krow + half * 16);        // d = 0..31 chunk
      v16h bk1 = load_b_f16(krow + 32 + half * 16);   // d = 32..63 chunk
      v8f z;
#pragma unroll
      for (int r = 0; r < 8; ++r) z[r] = 0.0f;
      z = wmma16(aq0, bk0, z);
      sc[nb] = wmma16(aq1, bk1, z);
    }

    // ---- causal mask + online softmax (row = r + 8*half across 16 lanes) ----
#pragma unroll
    for (int r = 0; r < 8; ++r) {
      const int qi = q0 + r + 8 * half;
      const int k0i = j0 + ln;
      const int k1i = j0 + 16 + ln;
      float s0 = (k0i <= qi) ? sc[0][r] * scale : -1.0e9f;
      float s1 = (k1i <= qi) ? sc[1][r] * scale : -1.0e9f;

      float t = fmaxf(s0, s1);
#pragma unroll
      for (int mk = 1; mk < 16; mk <<= 1) t = fmaxf(t, __shfl_xor(t, mk, 32));
      const float mnew  = fmaxf(mi[r], t);
      const float alpha = __expf(mi[r] - mnew);
      const float p0 = __expf(s0 - mnew);
      const float p1 = __expf(s1 - mnew);
      float rs = p0 + p1;
#pragma unroll
      for (int mk = 1; mk < 16; mk <<= 1) rs += __shfl_xor(rs, mk, 32);
      li[r] = li[r] * alpha + rs;
      mi[r] = mnew;
#pragma unroll
      for (int nb = 0; nb < 4; ++nb) o[nb][r] *= alpha;

      const int mrow = r + 8 * half;
      lds_p[wid][mrow * 32 + ln]      = (h16)p0;
      lds_p[wid][mrow * 32 + 16 + ln] = (h16)p1;
    }

    // wave-private LDS round trip: drain DS stores before A-layout reload
    asm volatile("s_wait_dscnt 0" ::: "memory");

    // reload P as A-fragment (16x32, K = key offset)
    v16h ap;
#pragma unroll
    for (int v = 0; v < 8; ++v) {
      const int k = a_koff(v, half);
      ap[2 * v]     = lds_p[wid][ln * 32 + k];
      ap[2 * v + 1] = lds_p[wid][ln * 32 + k + 1];
    }

    // ---- O += P @ V : 4 N-blocks of 16 dims, K=32 keys in one WMMA ----
#pragma unroll
    for (int nb = 0; nb < 4; ++nb) {
      const h16* vrow = vbase + (size_t)(nb * 16 + ln) * SS + j0 + half * 16;
      v16h bv = load_b_f16(vrow);
      o[nb] = wmma16(ap, bv, o[nb]);
    }
  }

  // ---- normalize and store O as [B, S, D] f16 ----
#pragma unroll
  for (int r = 0; r < 8; ++r) {
    const float inv = 1.0f / li[r];
    const int srow = q0 + r + 8 * half;
#pragma unroll
    for (int nb = 0; nb < 4; ++nb) {
      Oflat[((size_t)bi * SS + srow) * DM + hh * DKH + nb * 16 + ln] =
          (h16)(o[nb][r] * inv);
    }
  }
}

// ====================================================================
// Kernel 3: output projection  out = O @ Wo^T + bo  (f16 in, f32 out)
// ====================================================================
__global__ __launch_bounds__(256) void mha_oproj_kernel(
    const h16* __restrict__ X, const h16* __restrict__ W,
    const float* __restrict__ bias, float* __restrict__ Out)
{
  const int wid  = threadIdx.x >> 5;
  const int lane = threadIdx.x & 31;
  const int gw   = blockIdx.x * 8 + wid;
  const int tileN = gw & 15;
  const int tileM = gw >> 4;
  const int half = lane >> 4;
  const int ln   = lane & 15;

  const h16* xrow = X + (size_t)(tileM * 16 + ln) * DM;

  v8f acc[4];
#pragma unroll
  for (int i = 0; i < 4; ++i)
#pragma unroll
    for (int r = 0; r < 8; ++r) acc[i][r] = 0.0f;

  for (int k0 = 0; k0 < DM; k0 += 32) {
    __builtin_prefetch(xrow + k0 + 64, 0, 0);
    v16h a = load_a_f16(xrow, k0, half);
#pragma unroll
    for (int nb = 0; nb < 4; ++nb) {
      const int wrow = tileN * 64 + nb * 16 + ln;
      v16h b = load_b_f16(W + (size_t)wrow * DM + k0 + half * 16);
      acc[nb] = wmma16(a, b, acc[nb]);
    }
  }

#pragma unroll
  for (int nb = 0; nb < 4; ++nb) {
    const int ncol = tileN * 64 + nb * 16 + ln;
    const float bv = bias[ncol];
#pragma unroll
    for (int r = 0; r < 8; ++r) {
      const int gm = tileM * 16 + r + 8 * half;
      Out[(size_t)gm * DM + ncol] = acc[nb][r] + bv;
    }
  }
}

// ====================================================================
extern "C" void kernel_launch(void* const* d_in, const int* in_sizes, int n_in,
                              void* d_out, int out_size, void* d_ws, size_t ws_size,
                              hipStream_t stream) {
  const float* q    = (const float*)d_in[0];
  const float* k    = (const float*)d_in[1];
  const float* v    = (const float*)d_in[2];
  // d_in[3] = causal mask (computed analytically in-kernel)
  const float* wq   = (const float*)d_in[4];
  const float* bq   = (const float*)d_in[5];
  const float* wk   = (const float*)d_in[6];
  const float* bk   = (const float*)d_in[7];
  const float* wv   = (const float*)d_in[8];
  const float* bv   = (const float*)d_in[9];
  const float* wo   = (const float*)d_in[10];
  const float* bo   = (const float*)d_in[11];
  float* out = (float*)d_out;

  const size_t xel = (size_t)MTOT * DM;            // 4M elements (X / head bufs)
  const size_t wel = (size_t)DM * DM;              // 1M elements (weights)
  h16* Qh  = (h16*)d_ws;          // [B,H,S,dk]
  h16* Kh  = Qh  + xel;           // [B,H,S,dk]
  h16* Vt  = Kh  + xel;           // [B,H,dk,S]
  h16* Of  = Vt  + xel;           // [B,S,D]
  h16* Xq  = Of  + xel;           // f16 copies of inputs
  h16* Xk  = Xq  + xel;
  h16* Xv  = Xk  + xel;
  h16* Wq  = Xv  + xel;           // f16 copies of weights
  h16* Wk  = Wq  + wel;
  h16* Wv  = Wk  + wel;
  h16* Wo  = Wv  + wel;

  // --- one-shot fp32 -> f16 conversions (streaming, HBM-bound, ~3us) ---
  const int xblk = (int)(xel / 8 / 256);   // 2048
  const int wblk = (int)(wel / 8 / 256);   // 512
  cvt_f32_f16_kernel<<<xblk, 256, 0, stream>>>(q,  Xq, (int)(xel / 8));
  cvt_f32_f16_kernel<<<xblk, 256, 0, stream>>>(k,  Xk, (int)(xel / 8));
  cvt_f32_f16_kernel<<<xblk, 256, 0, stream>>>(v,  Xv, (int)(xel / 8));
  cvt_f32_f16_kernel<<<wblk, 256, 0, stream>>>(wq, Wq, (int)(wel / 8));
  cvt_f32_f16_kernel<<<wblk, 256, 0, stream>>>(wk, Wk, (int)(wel / 8));
  cvt_f32_f16_kernel<<<wblk, 256, 0, stream>>>(wv, Wv, (int)(wel / 8));
  cvt_f32_f16_kernel<<<wblk, 256, 0, stream>>>(wo, Wo, (int)(wel / 8));

  const int nblk = (MTOT / 16) * (DM / 64) / 8;   // 512 blocks, 8 waves each

  mha_proj_kernel<<<nblk, 256, 0, stream>>>(Xq, Wq, bq, Qh, 0);
  mha_proj_kernel<<<nblk, 256, 0, stream>>>(Xk, Wk, bk, Kh, 0);
  mha_proj_kernel<<<nblk, 256, 0, stream>>>(Xv, Wv, bv, Vt, 1);
  mha_attn_kernel<<<nblk, 256, 0, stream>>>(Qh, Kh, Vt, Of);
  mha_oproj_kernel<<<nblk, 256, 0, stream>>>(Of, Wo, bo, out);
}
